// GraphSAGEModel_9483287789910
// MI455X (gfx1250) — matrix operand order repaired and checked by reading, not compiled, not run on hardware
//
#include <hip/hip_runtime.h>
#include <hip/hip_bf16.h>
#include <stdint.h>

#define NN      100000
#define NE      1600000
#define IN_DIM  160
#define HID     128
#define BN_EPS  1e-5f

typedef __attribute__((ext_vector_type(16))) __bf16 v16bf;
typedef __attribute__((ext_vector_type(8)))  float  v8f;

// round-to-nearest-even f32 -> bf16 (bit pattern as u16), for weight packing
static __device__ __forceinline__ unsigned short f2bf(float f) {
    unsigned u = __float_as_uint(f);
    u += 0x7FFFu + ((u >> 16) & 1u);
    return (unsigned short)(u >> 16);
}

__global__ void k_zero(float* __restrict__ p, int n) {
    int i = blockIdx.x * blockDim.x + threadIdx.x;
    if (i < n) p[i] = 0.0f;
}

__global__ void k_degree(const long long* __restrict__ dst, float* __restrict__ cnt, int E) {
    int e = blockIdx.x * blockDim.x + threadIdx.x;
    if (e < E) atomicAdd(&cnt[dst[e]], 1.0f);
}

__global__ void k_inv(const float* __restrict__ cnt, float* __restrict__ inv, int n) {
    int i = blockIdx.x * blockDim.x + threadIdx.x;
    if (i < n) inv[i] = 1.0f / fmaxf(cnt[i], 1.0f);
}

// Pack W[d x 128] (row-major f32) -> Wt[128 x d] (bf16, column of W contiguous).
// This is exactly the B-fragment order: lane reads Wt[col*d + k0 + khi*16 .. +16].
__global__ void k_packw(const float* __restrict__ w, unsigned short* __restrict__ wt, int d) {
    int i = blockIdx.x * blockDim.x + threadIdx.x;
    if (i < HID * d) {
        int col = i / d, k = i - col * d;
        wt[i] = f2bf(w[(size_t)k * HID + col]);
    }
}

// one wave per edge; each lane moves 4 consecutive features per iteration
__global__ void k_scatter(const long long* __restrict__ src, const long long* __restrict__ dst,
                          const float* __restrict__ h, float* __restrict__ agg,
                          int d, int E) {
    int e = blockIdx.x * 8 + (threadIdx.x >> 5);
    if (e >= E) return;
    int lane = threadIdx.x & 31;
    long long s = src[e];
    long long t = dst[e];
    const float* hp = h + (size_t)s * d;
    float*       ap = agg + (size_t)t * d;
    for (int f = lane * 4; f < d; f += 128) {
        float4 v = *(const float4*)(hp + f);
        atomicAdd(&ap[f + 0], v.x);
        atomicAdd(&ap[f + 1], v.y);
        atomicAdd(&ap[f + 2], v.z);
        atomicAdd(&ap[f + 3], v.w);
    }
}

// Fused SAGE layer: out[16 x 128] per block = (agg*inv) @ wl + h @ wr, +bias, BN, ReLU.
// 256 threads = 8 waves; wave w owns output columns [16w, 16w+16).
// A fragments loaded directly from global (two aligned 32B reads/lane/step);
// B fragments loaded directly from pre-packed bf16 weights (one 32B read/lane/step).
template <int D>
__global__ __launch_bounds__(256) void k_sage_gemm(
    const float* __restrict__ agg, const float* __restrict__ inv, const float* __restrict__ h,
    const unsigned short* __restrict__ wlT, const unsigned short* __restrict__ wrT,
    const float* __restrict__ bl,  const float* __restrict__ gam, const float* __restrict__ bet,
    const float* __restrict__ rm,  const float* __restrict__ rv,
    float* __restrict__ out)
{
    const int t    = threadIdx.x;
    const int wave = t >> 5;
    const int lane = t & 31;
    const int row0 = blockIdx.x * 16;
    const int m    = lane & 15;       // A row-in-tile / output column-in-slab
    const int khi  = lane >> 4;       // lane-half selector
    const int col  = wave * 16 + m;   // this lane's output column

    v8f acc = {};

#pragma unroll
    for (int phase = 0; phase < 2; ++phase) {
        const float* __restrict__ A = phase ? h : agg;
        const unsigned short* __restrict__ Wt = phase ? wrT : wlT;
        const float s = phase ? 1.0f : inv[row0 + m];           // folds to 1.0 in phase 1
        const float* __restrict__ ap = A + (size_t)(row0 + m) * D;
        const unsigned short* __restrict__ bp = Wt + (size_t)col * D + khi * 16;

#pragma unroll
        for (int k0 = 0; k0 < D; k0 += 32) {
            // A fragment: row m, K = k0+khi*16+{0..7} and k0+khi*16+8+{0..7}
            v8f f0 = *(const v8f*)(ap + k0 + khi * 16);
            v8f f1 = *(const v8f*)(ap + k0 + khi * 16 + 8);
            v16bf a;
#pragma unroll
            for (int j = 0; j < 8; ++j) a[j] = (__bf16)(f0[j] * s);
#pragma unroll
            for (int j = 0; j < 8; ++j) a[j + 8] = (__bf16)(f1[j] * s);
            // B fragment: column `col`, K = k0 + khi*16 + {0..15}, contiguous bf16
            v16bf b = *(const v16bf*)(bp + k0);

            acc = __builtin_amdgcn_wmma_f32_16x16x32_bf16(
                false, a, false, b, (short)0, acc, false, false);
        }
    }

    // epilogue: +bias, BatchNorm(running stats), ReLU; coalesced 64B row segments
    float blv   = bl[col];
    float rmv   = rm[col];
    float bev   = bet[col];
    float scale = gam[col] * rsqrtf(rv[col] + BN_EPS);
#pragma unroll
    for (int r = 0; r < 8; ++r) {
        int row = row0 + khi * 8 + r;   // C layout: VGPR r -> M = r + 8*khi
        float v = (acc[r] + blv - rmv) * scale + bev;
        out[(size_t)row * HID + col] = fmaxf(v, 0.0f);
    }
}

// final linear 128 -> 1: one wave per node, one float4 per lane, shuffle reduction
__global__ void k_final(const float* __restrict__ h, const float* __restrict__ w,
                        const float* __restrict__ b, float* __restrict__ out, int N) {
    int node = blockIdx.x * 8 + (threadIdx.x >> 5);
    if (node >= N) return;
    int lane = threadIdx.x & 31;
    const float4 hv = *(const float4*)(h + (size_t)node * HID + lane * 4);
    const float4 wv = *(const float4*)(w + lane * 4);
    float acc = hv.x * wv.x + hv.y * wv.y + hv.z * wv.z + hv.w * wv.w;
    for (int off = 16; off > 0; off >>= 1) acc += __shfl_xor(acc, off, 32);
    if (lane == 0) out[node] = acc + b[0];
}

extern "C" void kernel_launch(void* const* d_in, const int* in_sizes, int n_in,
                              void* d_out, int out_size, void* d_ws, size_t ws_size,
                              hipStream_t stream) {
    (void)in_sizes; (void)n_in; (void)out_size; (void)ws_size;
    const float*     x    = (const float*)d_in[0];
    const long long* ei   = (const long long*)d_in[1];
    const long long* src  = ei;
    const long long* dst  = ei + NE;
    const float*     linw = (const float*)d_in[23];
    const float*     linb = (const float*)d_in[24];

    // workspace carve-out
    char* ws = (char*)d_ws;
    size_t off = 0;
    auto carve = [&](size_t bytes) -> void* {
        void* p = ws + off;
        off = (off + bytes + 255) & ~(size_t)255;
        return p;
    };
    float* agg = (float*)carve((size_t)NN * IN_DIM * sizeof(float));
    float* h1  = (float*)carve((size_t)NN * HID * sizeof(float));
    float* h2  = (float*)carve((size_t)NN * HID * sizeof(float));
    float* cnt = (float*)carve((size_t)NN * sizeof(float));
    float* inv = (float*)carve((size_t)NN * sizeof(float));
    unsigned short* wT[6];
    int wdims[6] = { IN_DIM, IN_DIM, HID, HID, HID, HID };   // wl0, wr0, wl1, wr1, wl2, wr2
    for (int i = 0; i < 6; ++i)
        wT[i] = (unsigned short*)carve((size_t)HID * wdims[i] * sizeof(unsigned short));

    // pack weights -> bf16, B-fragment order (once per launch; tiny)
    int widx[6] = { 2, 3, 9, 10, 16, 17 };
    for (int i = 0; i < 6; ++i) {
        int n = HID * wdims[i];
        k_packw<<<(n + 255) / 256, 256, 0, stream>>>((const float*)d_in[widx[i]], wT[i], wdims[i]);
    }

    // in-degree (same for all layers)
    k_zero  <<<(NN + 255) / 256, 256, 0, stream>>>(cnt, NN);
    k_degree<<<(NE + 255) / 256, 256, 0, stream>>>(dst, cnt, NE);
    k_inv   <<<(NN + 255) / 256, 256, 0, stream>>>(cnt, inv, NN);

    const float* hin = x;
    float* houts[3] = { h1, h2, h1 };
    int    dims[3]  = { IN_DIM, HID, HID };
    for (int l = 0; l < 3; ++l) {
        int d  = dims[l];
        int na = NN * d;
        int base = 2 + 7 * l;  // wl, wr, bl, g, be, rm, rv
        k_zero   <<<(na + 255) / 256, 256, 0, stream>>>(agg, na);
        k_scatter<<<(NE + 7) / 8, 256, 0, stream>>>(src, dst, hin, agg, d, NE);
        if (d == IN_DIM) {
            k_sage_gemm<IN_DIM><<<NN / 16, 256, 0, stream>>>(
                agg, inv, hin, wT[2 * l], wT[2 * l + 1],
                (const float*)d_in[base + 2], (const float*)d_in[base + 3],
                (const float*)d_in[base + 4], (const float*)d_in[base + 5],
                (const float*)d_in[base + 6], houts[l]);
        } else {
            k_sage_gemm<HID><<<NN / 16, 256, 0, stream>>>(
                agg, inv, hin, wT[2 * l], wT[2 * l + 1],
                (const float*)d_in[base + 2], (const float*)d_in[base + 3],
                (const float*)d_in[base + 4], (const float*)d_in[base + 5],
                (const float*)d_in[base + 6], houts[l]);
        }
        hin = houts[l];
    }

    k_final<<<(NN + 7) / 8, 256, 0, stream>>>(hin, linw, linb, (float*)d_out, NN);
}
